// GraphSAGE_43705587204164
// MI455X (gfx1250) — matrix operand order repaired and checked by reading, not compiled, not run on hardware
//
#include <hip/hip_runtime.h>

#define N_NODES 100000
#define N_EDGES 1600000
#define EPSV 1e-5f

typedef __attribute__((ext_vector_type(16))) __bf16 v16bf;
typedef __attribute__((ext_vector_type(8)))  float  v8f;

union FragU { uint4 u4[2]; v16bf v; };
union AccU  { v8f v; float f[8]; };

__device__ __forceinline__ unsigned short f2bf(float f) {
  unsigned int u = __float_as_uint(f);
  unsigned int r = ((u >> 16) & 1u) + 0x7fffu;   // round-to-nearest-even
  return (unsigned short)((u + r) >> 16);
}

// ---------------------------------------------------------------- zero
__global__ void zero_f32(float* __restrict__ p, int n) {
  int i = blockIdx.x * blockDim.x + threadIdx.x;
  if (i < n) p[i] = 0.f;
}

// ---------------------------------------------------------------- weight prep
// Pack 8 matrices (Wl[0..3], Wr[0..3], each 128x128 f32 row-major K x N) into
// bf16 WMMA B-fragments. Per matrix: 4 K-tiles x 8 N-tiles, each frag =
// 32 lanes x 8 dwords. Lane<16: N = ntile*16+lane, K = kt*32 + (0..15);
// lane>=16: N = ntile*16+lane-16, K = kt*32 + 16 + (0..15). K pairs per dword.
__global__ void prep_frags(const float* __restrict__ Wl,
                           const float* __restrict__ Wr,
                           unsigned int* __restrict__ frags) {
  int t = blockIdx.x * 256 + threadIdx.x;          // 0 .. 8191
  int m    = t >> 10;                              // matrix 0..7
  int blk  = (t >> 5) & 31;                        // kt*8 + ntile
  int lane = t & 31;
  int kt = blk >> 3, ntile = blk & 7;
  const float* W = (m < 4) ? (Wl + (size_t)m * 128 * 128)
                           : (Wr + (size_t)(m - 4) * 128 * 128);
  int n = ntile * 16 + (lane & 15);
  int kbase = kt * 32 + (lane >> 4) * 16;
  unsigned int* dst = frags + ((size_t)m * 32 + blk) * 256 + (size_t)lane * 8;
#pragma unroll
  for (int j = 0; j < 8; ++j) {
    unsigned short lo = f2bf(W[(size_t)(kbase + 2 * j)     * 128 + n]);
    unsigned short hi = f2bf(W[(size_t)(kbase + 2 * j + 1) * 128 + n]);
    dst[j] = (unsigned int)lo | ((unsigned int)hi << 16);
  }
}

// ---------------------------------------------------------------- encoder MLP
__global__ void encoder_kernel(const float* __restrict__ x,
                               const float* __restrict__ W1, const float* __restrict__ b1,
                               const float* __restrict__ W2, const float* __restrict__ b2,
                               float* __restrict__ z, unsigned short* __restrict__ zbf) {
  __shared__ float h[2][64];
  int l = threadIdx.x >> 6;
  int t = threadIdx.x & 63;
  int node = blockIdx.x * 2 + l;
  if (node < N_NODES) {
    const float* xr = x + (size_t)node * 7;
    float acc = b1[t];
#pragma unroll
    for (int k = 0; k < 7; ++k) acc += xr[k] * W1[k * 64 + t];
    h[l][t] = fmaxf(acc, 0.f);
  }
  __syncthreads();
  if (node >= N_NODES) return;
#pragma unroll
  for (int half = 0; half < 2; ++half) {
    int c = t + half * 64;
    float acc = b2[c];
    for (int k = 0; k < 64; ++k) acc += h[l][k] * W2[k * 128 + c];
    z[(size_t)node * 128 + c]   = acc;
    zbf[(size_t)node * 128 + c] = f2bf(acc);
  }
}

// ---------------------------------------------------------------- edge scatter
// One wave per edge: 32 lanes x float4 gather from z[src], atomic f32 scatter
// into agg[dst]; lane 0 bumps degree. Prefetch source row 8 edges ahead to
// overlap the ei->row pointer-chase (global_prefetch_b8 path).
__global__ void scatter_kernel(const float* __restrict__ z,
                               const long long* __restrict__ ei,
                               float* __restrict__ agg, float* __restrict__ deg) {
  int wid  = (blockIdx.x * blockDim.x + threadIdx.x) >> 5;
  int lane = threadIdx.x & 31;
  if (wid >= N_EDGES) return;
  if (wid + 8 < N_EDGES) {
    int psrc = (int)ei[wid + 8];
    __builtin_prefetch(z + (size_t)psrc * 128 + lane * 4, 0, 1);
  }
  int src = (int)ei[wid];
  int dst = (int)ei[(size_t)N_EDGES + wid];
  float4 v = *(const float4*)(z + (size_t)src * 128 + lane * 4);
  float* a = agg + (size_t)dst * 128 + lane * 4;
  atomicAdd(a + 0, v.x);
  atomicAdd(a + 1, v.y);
  atomicAdd(a + 2, v.z);
  atomicAdd(a + 3, v.w);
  if (lane == 0) atomicAdd(deg + dst, 1.0f);
}

// ---------------------------------------------------------------- mean -> bf16
__global__ void mean_bf_kernel(const float* __restrict__ agg,
                               const float* __restrict__ deg,
                               unsigned short* __restrict__ meanbf) {
  int t = blockIdx.x * blockDim.x + threadIdx.x;
  if (t >= N_NODES * 32) return;
  int node = t >> 5, c4 = (t & 31) * 4;
  float inv = 1.0f / fmaxf(deg[node], 1.0f);
  float4 v = *(const float4*)(agg + (size_t)node * 128 + c4);
  ushort4 o;
  o.x = f2bf(v.x * inv); o.y = f2bf(v.y * inv);
  o.z = f2bf(v.z * inv); o.w = f2bf(v.w * inv);
  *(ushort4*)(meanbf + (size_t)node * 128 + c4) = o;
}

// ---------------------------------------------------------------- WMMA GEMM
// out[N,128] = meanbf @ Wl + zbf @ Wr + bl  (bf16 inputs, f32 accumulate)
// Block = 8 waves. All Wl+Wr fragments for the layer (64 KB) are staged into
// LDS once per block, then each wave computes a 16(M) x 64(N) tile:
// 4 K-steps x 4 N-tiles x 2 matmuls = 32 v_wmma per wave, B served by ds loads.
__global__ void sage_gemm_kernel(const unsigned short* __restrict__ meanbf,
                                 const unsigned short* __restrict__ zbf,
                                 const unsigned int* __restrict__ fragWl,
                                 const unsigned int* __restrict__ fragWr,
                                 const float* __restrict__ bl,
                                 float* __restrict__ out) {
  __shared__ unsigned int sfrag[16384];            // 64 KB: [0,8192)=Wl, [8192,16384)=Wr
  {
    const uint4* gl = (const uint4*)fragWl;        // 2048 uint4
    const uint4* gr = (const uint4*)fragWr;        // 2048 uint4
    uint4* sl = (uint4*)sfrag;
    uint4* sr = (uint4*)(sfrag + 8192);
    int t = threadIdx.x;
#pragma unroll
    for (int i = 0; i < 8; ++i) sl[t + i * 256] = gl[t + i * 256];
#pragma unroll
    for (int i = 0; i < 8; ++i) sr[t + i * 256] = gr[t + i * 256];
  }
  __syncthreads();

  const int nTasks = (N_NODES / 16) * 2;           // 12500
  int wave = threadIdx.x >> 5;
  int lane = threadIdx.x & 31;
  int task = blockIdx.x * 8 + wave;
  if (task >= nTasks) return;
  int mtile = task >> 1, nhalf = task & 1;
  int row0 = mtile * 16, col0 = nhalf * 64;

  // A-fragment addressing (contiguous K order, same as the packed B frags):
  // lane<16 -> row = row0+lane, K = k0..k0+15
  // lane>=16 -> row = row0+lane-16, K = k0+16..k0+31
  int rowA = row0 + (lane & 15);
  int khi  = (lane >> 4) * 16;
  const unsigned short* am = meanbf + (size_t)rowA * 128;
  const unsigned short* az = zbf    + (size_t)rowA * 128;

  v8f acc[4] = {};
#pragma unroll
  for (int kt = 0; kt < 4; ++kt) {
    int k0 = kt * 32;
    FragU aM, aZ;
    aM.u4[0] = *(const uint4*)(am + k0 + khi);
    aM.u4[1] = *(const uint4*)(am + k0 + khi + 8);
    aZ.u4[0] = *(const uint4*)(az + k0 + khi);
    aZ.u4[1] = *(const uint4*)(az + k0 + khi + 8);
#pragma unroll
    for (int nt = 0; nt < 4; ++nt) {
      int ntile = nhalf * 4 + nt;
      const uint4* pL = (const uint4*)(sfrag +        ((kt * 8 + ntile) * 32 + lane) * 8);
      const uint4* pR = (const uint4*)(sfrag + 8192 + ((kt * 8 + ntile) * 32 + lane) * 8);
      FragU bL, bR;
      bL.u4[0] = pL[0]; bL.u4[1] = pL[1];
      bR.u4[0] = pR[0]; bR.u4[1] = pR[1];
      acc[nt] = __builtin_amdgcn_wmma_f32_16x16x32_bf16(
          false, aM.v, false, bL.v, (short)0, acc[nt], false, false);
      acc[nt] = __builtin_amdgcn_wmma_f32_16x16x32_bf16(
          false, aZ.v, false, bR.v, (short)0, acc[nt], false, false);
    }
  }

  // C/D layout: VGPR r -> lanes 0-15: (M=r, N=lane); lanes 16-31: (M=r+8, N=lane-16)
  int rbase = row0 + (lane >> 4) * 8;
#pragma unroll
  for (int nt = 0; nt < 4; ++nt) {
    AccU cu; cu.v = acc[nt];
    int col = col0 + nt * 16 + (lane & 15);
    float bb = bl[col];
    float* o = out + (size_t)rbase * 128 + col;
#pragma unroll
    for (int r = 0; r < 8; ++r) o[(size_t)r * 128] = cu.f[r] + bb;
  }
}

// ---------------------------------------------------------------- BatchNorm
__global__ void bn_reduce_kernel(const float* __restrict__ out,
                                 float* __restrict__ bnsum, float* __restrict__ bnsq) {
  int c = threadIdx.x & 127;
  int r = blockIdx.x * 2 + (threadIdx.x >> 7);
  int stride = gridDim.x * 2;
  float s = 0.f, q = 0.f;
  for (; r < N_NODES; r += stride) {
    float v = out[(size_t)r * 128 + c];
    s += v; q += v * v;
  }
  atomicAdd(bnsum + c, s);
  atomicAdd(bnsq + c, q);
}

__global__ void bn_apply_kernel(const float* __restrict__ out,
                                const float* __restrict__ bnsum, const float* __restrict__ bnsq,
                                const float* __restrict__ g, const float* __restrict__ b,
                                float* __restrict__ z, unsigned short* __restrict__ zbf) {
  int t = blockIdx.x * blockDim.x + threadIdx.x;
  if (t >= N_NODES * 128) return;
  int c = t & 127;
  const float invN = 1.0f / (float)N_NODES;
  float mu  = bnsum[c] * invN;
  float var = bnsq[c] * invN - mu * mu;
  float v = (out[t] - mu) * rsqrtf(var + EPSV) * g[c] + b[c];
  v = fmaxf(v, 0.f);
  z[t]   = v;
  zbf[t] = f2bf(v);
}

// ---------------------------------------------------------------- decoder MLP
__global__ void decoder_kernel(const float* __restrict__ zin,
                               const float* __restrict__ W1, const float* __restrict__ b1,
                               const float* __restrict__ W2, const float* __restrict__ b2,
                               float* __restrict__ y) {
  __shared__ float zr[2][128];
  __shared__ float h[2][64];
  int l = threadIdx.x >> 6;
  int t = threadIdx.x & 63;
  int node = blockIdx.x * 2 + l;
  if (node < N_NODES) {
    zr[l][t]      = zin[(size_t)node * 128 + t];
    zr[l][t + 64] = zin[(size_t)node * 128 + t + 64];
  }
  __syncthreads();
  if (node < N_NODES) {
    float acc = b1[t];
    for (int k = 0; k < 128; ++k) acc += zr[l][k] * W1[k * 64 + t];
    h[l][t] = fmaxf(acc, 0.f);
  }
  __syncthreads();
  if (node < N_NODES && t < 4) {
    float acc = b2[t];
    for (int k = 0; k < 64; ++k) acc += h[l][k] * W2[k * 4 + t];
    y[(size_t)node * 4 + t] = acc;
  }
}

// ================================================================ launch
extern "C" void kernel_launch(void* const* d_in, const int* in_sizes, int n_in,
                              void* d_out, int out_size, void* d_ws, size_t ws_size,
                              hipStream_t stream) {
  const float*      x      = (const float*)d_in[0];
  const long long*  ei     = (const long long*)d_in[1];   // int64 [2, E]
  const float* encW1 = (const float*)d_in[2];
  const float* encb1 = (const float*)d_in[3];
  const float* encW2 = (const float*)d_in[4];
  const float* encb2 = (const float*)d_in[5];
  const float* sageWl = (const float*)d_in[6];   // [4,128,128]
  const float* sagebl = (const float*)d_in[7];   // [4,128]
  const float* sageWr = (const float*)d_in[8];   // [4,128,128]
  const float* bng    = (const float*)d_in[9];   // [3,128]
  const float* bnb    = (const float*)d_in[10];  // [3,128]
  const float* decW1  = (const float*)d_in[11];
  const float* decb1  = (const float*)d_in[12];
  const float* decW2  = (const float*)d_in[13];
  const float* decb2  = (const float*)d_in[14];
  float* y = (float*)d_out;

  // workspace carve-up (256B aligned)
  char* ws = (char*)d_ws;
  size_t off = 0;
  auto carve = [&](size_t bytes) {
    size_t o = off;
    off += (bytes + 255) & ~(size_t)255;
    return (void*)(ws + o);
  };
  float*          AGG    = (float*)carve((size_t)N_NODES * 128 * 4); // agg sums, reused as GEMM out
  float*          Z      = (float*)carve((size_t)N_NODES * 128 * 4); // current features f32
  unsigned short* ZBF    = (unsigned short*)carve((size_t)N_NODES * 128 * 2);
  unsigned short* MEANBF = (unsigned short*)carve((size_t)N_NODES * 128 * 2);
  float*          DEG    = (float*)carve((size_t)N_NODES * 4);
  float*          BNS    = (float*)carve(256 * 4);                   // [0:128)=sum, [128:256)=sumsq
  unsigned int*   FRAGS  = (unsigned int*)carve((size_t)8 * 8192 * 4);
  (void)ws_size; (void)in_sizes; (void)n_in; (void)out_size;

  // 1) weight fragments (once per call — deterministic)
  prep_frags<<<32, 256, 0, stream>>>(sageWl, sageWr, FRAGS);

  // 2) encoder
  encoder_kernel<<<N_NODES / 2, 128, 0, stream>>>(x, encW1, encb1, encW2, encb2, Z, ZBF);

  const int gemmBlocks = ((N_NODES / 16) * 2 + 7) / 8;   // 1563
  // 3) four SAGE layers
  for (int i = 0; i < 4; ++i) {
    zero_f32<<<(N_NODES * 128 + 255) / 256, 256, 0, stream>>>(AGG, N_NODES * 128);
    zero_f32<<<(N_NODES + 255) / 256, 256, 0, stream>>>(DEG, N_NODES);
    if (i < 3) zero_f32<<<1, 256, 0, stream>>>(BNS, 256);

    scatter_kernel<<<(N_EDGES * 32) / 256, 256, 0, stream>>>(Z, ei, AGG, DEG);
    mean_bf_kernel<<<(N_NODES * 32 + 255) / 256, 256, 0, stream>>>(AGG, DEG, MEANBF);

    sage_gemm_kernel<<<gemmBlocks, 256, 0, stream>>>(
        MEANBF, ZBF,
        FRAGS + (size_t)i * 8192,            // Wl frags
        FRAGS + (size_t)(4 + i) * 8192,      // Wr frags
        sagebl + (size_t)i * 128,
        AGG);                                // reuse AGG as GEMM output

    if (i < 3) {
      bn_reduce_kernel<<<512, 256, 0, stream>>>(AGG, BNS, BNS + 128);
      bn_apply_kernel<<<(N_NODES * 128 + 255) / 256, 256, 0, stream>>>(
          AGG, BNS, BNS + 128, bng + (size_t)i * 128, bnb + (size_t)i * 128, Z, ZBF);
    }
  }

  // 4) decoder on layer-3 output (in AGG)
  decoder_kernel<<<N_NODES / 2, 128, 0, stream>>>(AGG, decW1, decb1, decW2, decb2, y);
}
